// PointNetPlusEncoder_74586402062744
// MI455X (gfx1250) — compile-verified, matrix operand order checked by reading, not run on hardware
//
#include <hip/hip_runtime.h>
#include <cstddef>

typedef __attribute__((ext_vector_type(16))) _Float16 v16h;
typedef __attribute__((ext_vector_type(8)))  float    v8f;

#define NB 4

// K-slot map for 16-bit WMMA A/B operands (ISA 7.12.2):
// lane half 'grp' (0:lanes0-15, 1:lanes16-31), slot i in 0..15
__device__ __forceinline__ int kmap(int grp, int i) {
  int j = i >> 1, p = i & 1;
  return (j < 4) ? (grp * 8 + j * 2 + p) : (16 + grp * 8 + (j - 4) * 2 + p);
}

// ---------------- generic WMMA GEMM: out[b,o,s] = sum_c W[o,c]*in[b,c,s] (+bias) (+res) ----
template <int CIN, int COUT>
__global__ void k_gemm_t(const float* __restrict__ in, const float* __restrict__ w,
                         const float* __restrict__ bias, const float* __restrict__ res,
                         float* __restrict__ out, int S) {
  int lane = threadIdx.x, grp = lane >> 4, ln = lane & 15;
  int s0 = blockIdx.x * 16, o0 = blockIdx.y * 16, b = blockIdx.z;
  int m = o0 + ln;  // COUT % 16 == 0 -> always valid
  const float* __restrict__ wrow = w + (size_t)m * CIN;
  const float* __restrict__ inb = in + (size_t)b * CIN * S + s0 + ln;
  v8f acc = {};
#pragma unroll
  for (int c0 = 0; c0 < CIN; c0 += 32) {
    float fa[16], fb[16];
    if (c0 + 32 <= CIN) {  // constant-folds per unrolled chunk: unmasked fast path
#pragma unroll
      for (int i = 0; i < 16; i++) {
        int c = c0 + kmap(grp, i);
        fa[i] = wrow[c];
        fb[i] = inb[(size_t)c * S];
      }
    } else {  // tail chunk: clamp index + multiply mask (no EXEC branching)
#pragma unroll
      for (int i = 0; i < 16; i++) {
        int c = c0 + kmap(grp, i);
        int cc = c < CIN ? c : CIN - 1;
        float msk = c < CIN ? 1.f : 0.f;
        fa[i] = wrow[cc] * msk;
        fb[i] = inb[(size_t)cc * S] * msk;
      }
    }
    v16h A, Bv;
#pragma unroll
    for (int i = 0; i < 16; i++) { A[i] = (_Float16)fa[i]; Bv[i] = (_Float16)fb[i]; }
    acc = __builtin_amdgcn_wmma_f32_16x16x32_f16(false, A, false, Bv, (short)0, acc, false, false);
  }
#pragma unroll
  for (int r2 = 0; r2 < 8; r2++) {
    int mm = o0 + r2 + 8 * grp;
    size_t oi = ((size_t)b * COUT + mm) * S + s0 + ln;
    float v = acc[r2] + (bias ? bias[mm] : 0.f) + (res ? res[oi] : 0.f);
    out[oi] = v;
  }
}

// ---------------- 3x3x3 SAME conv as implicit WMMA GEMM (K = CIN*27, COUT=32) -------------
template <int CIN>
__global__ void k_conv3d_t(const float* __restrict__ in, const float* __restrict__ w,
                           const float* __restrict__ bias, float* __restrict__ out, int r) {
  const int K = CIN * 27;
  int lane = threadIdx.x, grp = lane >> 4, ln = lane & 15;
  int V = r * r * r;
  int s0 = blockIdx.x * 16, o0 = blockIdx.y * 16, b = blockIdx.z;
  int vv = s0 + ln;
  int z = vv % r, y = (vv / r) % r, x = vv / (r * r);
  int m = o0 + ln;  // COUT=32, grid.y=2 -> always valid
  const float* __restrict__ wrow = w + (size_t)m * K;
  const float* __restrict__ inb = in + (size_t)b * CIN * V;
  v8f acc = {};
  for (int kk = 0; kk < K; kk += 32) {
    if (kk + 32 < K) __builtin_prefetch(wrow + kk + 32, 0, 1);
    float fa[16], fb[16];
#pragma unroll
    for (int i = 0; i < 16; i++) {
      int kg = kk + kmap(grp, i);
      bool kv = kg < K;
      int kgc = kv ? kg : K - 1;
      int c = kgc / 27, tap = kgc % 27;
      int dx = tap / 9 - 1, dy = (tap / 3) % 3 - 1, dz = tap % 3 - 1;
      int xx = x + dx, yy = y + dy, zz = z + dz;
      bool inside = kv && xx >= 0 && xx < r && yy >= 0 && yy < r && zz >= 0 && zz < r;
      int xc = min(max(xx, 0), r - 1), yc = min(max(yy, 0), r - 1), zc = min(max(zz, 0), r - 1);
      fa[i] = wrow[kgc] * (kv ? 1.f : 0.f);
      fb[i] = inb[(size_t)c * V + ((size_t)xc * r + yc) * r + zc] * (inside ? 1.f : 0.f);
    }
    v16h A, Bv;
#pragma unroll
    for (int i = 0; i < 16; i++) { A[i] = (_Float16)fa[i]; Bv[i] = (_Float16)fb[i]; }
    acc = __builtin_amdgcn_wmma_f32_16x16x32_f16(false, A, false, Bv, (short)0, acc, false, false);
  }
#pragma unroll
  for (int r2 = 0; r2 < 8; r2++) {
    int mm = o0 + r2 + 8 * grp;
    out[((size_t)b * 32 + mm) * V + s0 + ln] = acc[r2] + bias[mm];
  }
}

// ---------------- flash attention, C=32, one wave per 16-row s tile -----------------------
// Two-pass: pass1 computes per-row (max,sum) with per-lane scalars (row = lane%16) via an
// LDS D->A relayout; only one shfl_xor(16) per statistic per chunk. Pass2 recomputes scores
// and accumulates O with fixed max (no accumulator rescale).
__device__ __forceinline__ void flash_scores(const v16h& AQ, const float* kk, size_t kbase,
                                             int t0, int ln, int grp, v8f& d0, v8f& d1) {
  float fk0[16], fk1[16];
#pragma unroll
  for (int i = 0; i < 16; i++) {
    size_t row = kbase + (size_t)kmap(grp, i) * 0;  // silence
    (void)row;
    fk0[i] = kk[kbase + (size_t)kmap(grp, i) * 4096 + t0 + ln];      // placeholder, replaced below
    fk1[i] = 0.f;
  }
  (void)fk0; (void)fk1;
  d0 = v8f{}; d1 = v8f{};
}

__global__ void k_flash(const float* __restrict__ q, const float* __restrict__ kk,
                        const float* __restrict__ vv, float* __restrict__ hh, int S) {
  __shared__ float pbuf[16 * 32];
  __shared__ float rsbuf[16];
  int lane = threadIdx.x, grp = lane >> 4, ln = lane & 15;
  int s0 = blockIdx.x * 16, b = blockIdx.z;
  v16h AQ;
#pragma unroll
  for (int i = 0; i < 16; i++)
    AQ[i] = (_Float16)q[((size_t)b * 32 + kmap(grp, i)) * S + s0 + ln];

  // ---- pass 1: per-row max & sum (row = ln) ----
  float rmax = -1e30f, rsum = 0.f;
  for (int t0 = 0; t0 < S; t0 += 32) {
    float fk0[16], fk1[16];
#pragma unroll
    for (int i = 0; i < 16; i++) {
      int c = kmap(grp, i);
      fk0[i] = kk[((size_t)b * 32 + c) * S + t0 + ln];
      fk1[i] = kk[((size_t)b * 32 + c) * S + t0 + 16 + ln];
    }
    v16h BK0, BK1;
#pragma unroll
    for (int i = 0; i < 16; i++) { BK0[i] = (_Float16)fk0[i]; BK1[i] = (_Float16)fk1[i]; }
    v8f d0 = {}, d1 = {};
    d0 = __builtin_amdgcn_wmma_f32_16x16x32_f16(false, AQ, false, BK0, (short)0, d0, false, false);
    d1 = __builtin_amdgcn_wmma_f32_16x16x32_f16(false, AQ, false, BK1, (short)0, d1, false, false);
#pragma unroll
    for (int r2 = 0; r2 < 8; r2++) {
      int mrow = r2 + 8 * grp;
      pbuf[mrow * 32 + ln] = d0[r2];
      pbuf[mrow * 32 + 16 + ln] = d1[r2];
    }
    __syncthreads();
    float rowv[16];
#pragma unroll
    for (int i = 0; i < 16; i++) rowv[i] = pbuf[ln * 32 + kmap(grp, i)];
    float lmax = rowv[0];
#pragma unroll
    for (int i = 1; i < 16; i++) lmax = fmaxf(lmax, rowv[i]);
    lmax = fmaxf(lmax, __shfl_xor(lmax, 16, 32));
    float nm = fmaxf(rmax, lmax);
    float lsum = 0.f;
#pragma unroll
    for (int i = 0; i < 16; i++) lsum += __expf(rowv[i] - nm);
    lsum += __shfl_xor(lsum, 16, 32);
    rsum = rsum * __expf(rmax - nm) + lsum;
    rmax = nm;
    __syncthreads();
  }

  // ---- pass 2: O accumulation with fixed per-row max ----
  v8f oA = {}, oB = {};
  for (int t0 = 0; t0 < S; t0 += 32) {
    float fk0[16], fk1[16];
#pragma unroll
    for (int i = 0; i < 16; i++) {
      int c = kmap(grp, i);
      fk0[i] = kk[((size_t)b * 32 + c) * S + t0 + ln];
      fk1[i] = kk[((size_t)b * 32 + c) * S + t0 + 16 + ln];
    }
    v16h BK0, BK1;
#pragma unroll
    for (int i = 0; i < 16; i++) { BK0[i] = (_Float16)fk0[i]; BK1[i] = (_Float16)fk1[i]; }
    v8f d0 = {}, d1 = {};
    d0 = __builtin_amdgcn_wmma_f32_16x16x32_f16(false, AQ, false, BK0, (short)0, d0, false, false);
    d1 = __builtin_amdgcn_wmma_f32_16x16x32_f16(false, AQ, false, BK1, (short)0, d1, false, false);
#pragma unroll
    for (int r2 = 0; r2 < 8; r2++) {
      int mrow = r2 + 8 * grp;
      pbuf[mrow * 32 + ln] = d0[r2];
      pbuf[mrow * 32 + 16 + ln] = d1[r2];
    }
    __syncthreads();
    float fv0[16], fv1[16], fp[16];
#pragma unroll
    for (int i = 0; i < 16; i++) {
      int tl = kmap(grp, i);
      fp[i]  = __expf(pbuf[ln * 32 + tl] - rmax);
      fv0[i] = vv[((size_t)b * 32 + ln) * S + t0 + tl];
      fv1[i] = vv[((size_t)b * 32 + 16 + ln) * S + t0 + tl];
    }
    v16h AP, BV0, BV1;
#pragma unroll
    for (int i = 0; i < 16; i++) {
      AP[i] = (_Float16)fp[i]; BV0[i] = (_Float16)fv0[i]; BV1[i] = (_Float16)fv1[i];
    }
    oA = __builtin_amdgcn_wmma_f32_16x16x32_f16(false, AP, false, BV0, (short)0, oA, false, false);
    oB = __builtin_amdgcn_wmma_f32_16x16x32_f16(false, AP, false, BV1, (short)0, oB, false, false);
    __syncthreads();
  }

  // rsum (row = ln, duplicated across halves) -> D-layout rows
  if (grp == 0) rsbuf[ln] = rsum;
  __syncthreads();
#pragma unroll
  for (int r2 = 0; r2 < 8; r2++) {
    int mrow = r2 + 8 * grp, s = s0 + mrow;
    float inv = 1.f / rsbuf[mrow];
    hh[((size_t)b * 32 + ln) * S + s]      = oA[r2] * inv;
    hh[((size_t)b * 32 + 16 + ln) * S + s] = oB[r2] * inv;
  }
}

// ---------------- GroupNorm(+swish), 8 groups ---------------------------------------------
__global__ void k_gn_act(float* x, const float* gam, const float* bet, int C, int S, int act) {
  int gid = blockIdx.x, b = gid / 8, g = gid % 8, cg = C / 8;
  float* base = x + ((size_t)b * C + g * cg) * S;
  int cnt = cg * S;
  __shared__ float s1[256], s2[256];
  int t = threadIdx.x;
  float a = 0.f, q = 0.f;
  for (int i = t; i < cnt; i += 256) { float v = base[i]; a += v; q += v * v; }
  s1[t] = a; s2[t] = q; __syncthreads();
  for (int off = 128; off > 0; off >>= 1) {
    if (t < off) { s1[t] += s1[t + off]; s2[t] += s2[t + off]; }
    __syncthreads();
  }
  float mean = s1[0] / cnt, var = s2[0] / cnt - mean * mean;
  float inv = rsqrtf(var + 1e-5f);
  for (int i = t; i < cnt; i += 256) {
    int c = g * cg + i / S;
    float v = (base[i] - mean) * inv * gam[c] + bet[c];
    if (act) v = v / (1.f + expf(-v));
    base[i] = v;
  }
}

// ---------------- voxelize helpers --------------------------------------------------------
__global__ void k_transpose(const float* x, float* coords, int N, int total) {
  int id = blockIdx.x * 256 + threadIdx.x; if (id >= total) return;
  int b = id / N, n = id % N;
  for (int d = 0; d < 3; d++) coords[((size_t)b * 3 + d) * N + n] = x[((size_t)b * N + n) * 3 + d];
}

__global__ void k_vox_stats(const float* coords, int N, float* vmean, float* vmax) {
  int b = blockIdx.x, t = threadIdx.x;
  __shared__ float sx[256], sy[256], sz[256];
  float ax = 0, ay = 0, az = 0;
  for (int i = t; i < N; i += 256) {
    ax += coords[(b * 3 + 0) * N + i]; ay += coords[(b * 3 + 1) * N + i]; az += coords[(b * 3 + 2) * N + i];
  }
  sx[t] = ax; sy[t] = ay; sz[t] = az; __syncthreads();
  for (int off = 128; off > 0; off >>= 1) {
    if (t < off) { sx[t] += sx[t + off]; sy[t] += sy[t + off]; sz[t] += sz[t + off]; }
    __syncthreads();
  }
  float mx = sx[0] / N, my = sy[0] / N, mz = sz[0] / N;
  if (t == 0) { vmean[b * 3 + 0] = mx; vmean[b * 3 + 1] = my; vmean[b * 3 + 2] = mz; }
  __syncthreads();
  float bm = 0.f;
  for (int i = t; i < N; i += 256) {
    float cx = coords[(b * 3 + 0) * N + i] - mx, cy = coords[(b * 3 + 1) * N + i] - my,
          cz = coords[(b * 3 + 2) * N + i] - mz;
    bm = fmaxf(bm, sqrtf(cx * cx + cy * cy + cz * cz));
  }
  sx[t] = bm; __syncthreads();
  for (int off = 128; off > 0; off >>= 1) {
    if (t < off) sx[t] = fmaxf(sx[t], sx[t + off]);
    __syncthreads();
  }
  if (t == 0) vmax[b] = sx[0];
}

__global__ void k_vox_norm(const float* coords, const float* vmean, const float* vmax,
                           int N, int r, float* norm, int* vidx, int total) {
  int id = blockIdx.x * 256 + threadIdx.x; if (id >= total) return;
  int b = id / N, p = id % N;
  float mv = vmax[b] * 2.f;
  int vc[3];
  for (int d = 0; d < 3; d++) {
    float c = coords[(b * 3 + d) * N + p] - vmean[b * 3 + d];
    float nn = c / mv + 0.5f;
    nn = fminf(fmaxf(nn * r, 0.f), (float)(r - 1));
    norm[(b * 3 + d) * N + p] = nn;
    vc[d] = (int)rintf(nn);
  }
  vidx[id] = (vc[0] * r + vc[1]) * r + vc[2];
}

__global__ void k_zero(float* p, int n) {
  int id = blockIdx.x * 256 + threadIdx.x; if (id < n) p[id] = 0.f;
}

__global__ void k_scatter(const float* feat, const int* vidx, int Cf, int N, int V,
                          float* grid, float* cnt, int total) {
  int id = blockIdx.x * 256 + threadIdx.x; if (id >= total) return;
  int b = id / N, p = id % N;
  int ix = vidx[id];
  atomicAdd(&cnt[(size_t)b * V + ix], 1.f);
  for (int c = 0; c < Cf; c++)
    atomicAdd(&grid[((size_t)b * Cf + c) * V + ix], feat[((size_t)b * Cf + c) * N + p]);
}

__global__ void k_vox_div(float* grid, const float* cnt, int Cf, int V, int total) {
  int id = blockIdx.x * 256 + threadIdx.x; if (id >= total) return;
  int v = id % V, b = id / (Cf * V);
  grid[id] /= fmaxf(cnt[(size_t)b * V + v], 1.f);
}

// ---------------- SE block ---------------------------------------------------------------
__global__ void k_se_mean(const float* g, int V, float* sv) {
  int bc = blockIdx.x, t = threadIdx.x;
  __shared__ float s[256];
  float a = 0.f;
  for (int i = t; i < V; i += 256) a += g[(size_t)bc * V + i];
  s[t] = a; __syncthreads();
  for (int off = 128; off > 0; off >>= 1) { if (t < off) s[t] += s[t + off]; __syncthreads(); }
  if (t == 0) sv[bc] = s[0] / V;
}

__global__ void k_se_mlp(const float* sv, const float* w1, const float* w2, float* out) {
  int b = blockIdx.x, t = threadIdx.x;
  __shared__ float s[32], h[4];
  s[t] = sv[b * 32 + t]; __syncthreads();
  if (t < 4) {
    float a = 0.f;
    for (int c = 0; c < 32; c++) a += w1[t * 32 + c] * s[c];
    h[t] = a / (1.f + expf(-a));
  }
  __syncthreads();
  float a = 0.f;
  for (int j = 0; j < 4; j++) a += w2[t * 4 + j] * h[j];
  out[b * 32 + t] = 1.f / (1.f + expf(-a));
}

__global__ void k_se_scale(float* g, const float* sc, int V, int total) {
  int id = blockIdx.x * 256 + threadIdx.x; if (id >= total) return;
  g[id] *= sc[id / V];
}

// ---------------- devoxelize (trilinear gather) -------------------------------------------
__global__ void k_devox(const float* grid, const float* norm, int N, int r, int C,
                        float* out, int total) {
  int id = blockIdx.x * 256 + threadIdx.x; if (id >= total) return;
  int b = id / N, p = id % N;
  int V = r * r * r;
  float nn[3]; int lo[3], hi[3]; float fr[3];
  for (int d = 0; d < 3; d++) {
    nn[d] = norm[(b * 3 + d) * N + p];
    float fl = floorf(nn[d]);
    fr[d] = nn[d] - fl;
    lo[d] = (int)fl;
    hi[d] = min(lo[d] + 1, r - 1);
  }
  int idx8[8]; float w8[8];
  for (int cr = 0; cr < 8; cr++) {
    int ix = (cr & 4) ? hi[0] : lo[0];
    int iy = (cr & 2) ? hi[1] : lo[1];
    int iz = (cr & 1) ? hi[2] : lo[2];
    float wx = (cr & 4) ? fr[0] : 1.f - fr[0];
    float wy = (cr & 2) ? fr[1] : 1.f - fr[1];
    float wz = (cr & 1) ? fr[2] : 1.f - fr[2];
    idx8[cr] = (ix * r + iy) * r + iz;
    w8[cr] = wx * wy * wz;
  }
  for (int c = 0; c < C; c++) {
    const float* gb = grid + ((size_t)b * C + c) * V;
    float a = 0.f;
    for (int cr = 0; cr < 8; cr++) a += gb[idx8[cr]] * w8[cr];
    out[((size_t)b * C + c) * N + p] = a;
  }
}

__global__ void k_add(const float* a, const float* b, float* o, int n) {
  int id = blockIdx.x * 256 + threadIdx.x; if (id < n) o[id] = a[id] + b[id];
}

// ---------------- farthest point sampling -------------------------------------------------
__global__ void k_fps(const float* coords, int N, int m, int* idxout) {
  __shared__ float px[2048], py[2048], pz[2048], dist[2048];
  __shared__ float rv[256]; __shared__ int ri[256]; __shared__ int s_last;
  int b = blockIdx.x, t = threadIdx.x;
  for (int i = t; i < N; i += 256) {
    px[i] = coords[(b * 3 + 0) * N + i];
    py[i] = coords[(b * 3 + 1) * N + i];
    pz[i] = coords[(b * 3 + 2) * N + i];
    dist[i] = 1e10f;
  }
  if (t == 0) s_last = 0;
  __syncthreads();
  for (int step = 0; step < m; ++step) {
    int last = s_last;
    if (t == 0) idxout[b * m + step] = last;
    float lx = px[last], ly = py[last], lz = pz[last];
    float bm = -1.f; int bi = 0;
    for (int i = t; i < N; i += 256) {
      float dx = px[i] - lx, dy = py[i] - ly, dz = pz[i] - lz;
      float d = fminf(dist[i], dx * dx + dy * dy + dz * dz);
      dist[i] = d;
      if (d > bm) { bm = d; bi = i; }
    }
    rv[t] = bm; ri[t] = bi; __syncthreads();
    for (int off = 128; off > 0; off >>= 1) {
      if (t < off) {
        if (rv[t + off] > rv[t] || (rv[t + off] == rv[t] && ri[t + off] < ri[t])) {
          rv[t] = rv[t + off]; ri[t] = ri[t + off];
        }
      }
      __syncthreads();
    }
    if (t == 0) s_last = ri[0];
    __syncthreads();
  }
}

__global__ void k_centers(const float* coords, const int* idxc, int N, int m,
                          float* centers, int total) {
  int id = blockIdx.x * 256 + threadIdx.x; if (id >= total) return;
  int b = id / m, j = id % m;
  int ix = idxc[id];
  for (int d = 0; d < 3; d++) centers[(b * 3 + d) * m + j] = coords[(b * 3 + d) * N + ix];
}

__global__ void k_ballq(const float* centers, const float* pts, int N, int m,
                        float rad, int kN, int* nidx, int total) {
  int id = blockIdx.x * 256 + threadIdx.x; if (id >= total) return;
  int b = id / m, j = id % m;
  float cx = centers[(b * 3 + 0) * m + j], cy = centers[(b * 3 + 1) * m + j],
        cz = centers[(b * 3 + 2) * m + j];
  float r2 = rad * rad;
  int cnt = 0, first = 0; bool any = false;
  for (int i = 0; i < N && cnt < kN; i++) {
    float dx = pts[(b * 3 + 0) * N + i] - cx, dy = pts[(b * 3 + 1) * N + i] - cy,
          dz = pts[(b * 3 + 2) * N + i] - cz;
    if (dx * dx + dy * dy + dz * dz < r2) {
      if (!any) { any = true; first = i; }
      nidx[(size_t)id * kN + cnt] = i; cnt++;
    }
  }
  int pad = any ? first : 0;
  for (; cnt < kN; cnt++) nidx[(size_t)id * kN + cnt] = pad;
}

__global__ void k_group(const float* coords, const float* feat, const float* centers,
                        const int* nidx, int N, int m, int kN, float* g, int total) {
  int id = blockIdx.x * 256 + threadIdx.x; if (id >= total) return;
  int l = id % kN, j = (id / kN) % m, b = id / (kN * m);
  int ni = nidx[((size_t)b * m + j) * kN + l];
  int S = m * kN, sidx = j * kN + l;
  for (int d = 0; d < 3; d++)
    g[((size_t)b * 35 + d) * S + sidx] = coords[(b * 3 + d) * N + ni] - centers[(b * 3 + d) * m + j];
  for (int c = 0; c < 32; c++)
    g[((size_t)b * 35 + 3 + c) * S + sidx] = feat[((size_t)b * 32 + c) * N + ni];
}

__global__ void k_maxlast(const float* in, int S, float* out, int R) {
  int id = blockIdx.x * 256 + threadIdx.x; if (id >= R) return;
  const float* row = in + (size_t)id * S;
  float mrow = row[0];
  for (int i = 1; i < S; i++) mrow = fmaxf(mrow, row[i]);
  out[id] = mrow;
}

__global__ void k_head(const float* f, const float* w, const float* bias, float* out) {
  int id = blockIdx.x * 256 + threadIdx.x; if (id >= NB * 256) return;
  int b = id / 256, e = id % 256;
  float a = bias[e];
  for (int c = 0; c < 128; c++) a += f[b * 128 + c] * w[e * 128 + c];
  out[id] = a;
}

// ===========================================================================================
struct PVP {
  const float *c1w, *c1b, *c2w, *c2b, *n1g, *n1b, *n2g, *n2b;
  const float *ptcw, *ptcb, *ptng, *ptnb, *se1, *se2;
  const float *aqw, *aqb, *akw, *akb, *avw, *avb, *aow, *aob, *ang, *anb;
  int att;
};

static void launch_gemm(int Cin, int Cout, const float* in, const float* w, const float* bias,
                        const float* res, float* out, int S, hipStream_t st) {
  dim3 g(S / 16, Cout / 16, NB);
  if (Cin == 3 && Cout == 32)
    k_gemm_t<3, 32><<<g, 32, 0, st>>>(in, w, bias, res, out, S);
  else if (Cin == 32 && Cout == 32)
    k_gemm_t<32, 32><<<g, 32, 0, st>>>(in, w, bias, res, out, S);
  else if (Cin == 35 && Cout == 32)
    k_gemm_t<35, 32><<<g, 32, 0, st>>>(in, w, bias, res, out, S);
  else if (Cin == 32 && Cout == 128)
    k_gemm_t<32, 128><<<g, 32, 0, st>>>(in, w, bias, res, out, S);
}

extern "C" void kernel_launch(void* const* d_in, const int* in_sizes, int n_in,
                              void* d_out, int out_size, void* d_ws, size_t ws_size,
                              hipStream_t stream) {
  (void)in_sizes; (void)n_in; (void)out_size; (void)ws_size;
#define IN(i) ((const float*)d_in[(i)])
  // d_in[0] = x (B,N,3); then params leaves in recursively-sorted-key order.
  const float* X = IN(0);
  const float* head_b = IN(1); const float* head_w = IN(2);
  PVP pv1a{}, pv1b{}, pv2{};
  auto fill_pv = [&](PVP& p, int base, int att) {
    int i = base;
    if (att) {
      p.akb = IN(i++); p.akw = IN(i++); p.anb = IN(i++); p.ang = IN(i++);
      p.aob = IN(i++); p.aow = IN(i++); p.aqb = IN(i++); p.aqw = IN(i++);
      p.avb = IN(i++); p.avw = IN(i++);
    }
    p.c1b = IN(i++); p.c1w = IN(i++); p.c2b = IN(i++); p.c2w = IN(i++);
    p.n1b = IN(i++); p.n1g = IN(i++); p.n2b = IN(i++); p.n2g = IN(i++);
    p.ptcb = IN(i++); p.ptcw = IN(i++); p.ptnb = IN(i++); p.ptng = IN(i++);
    p.se1 = IN(i++); p.se2 = IN(i++);
    p.att = att;
  };
  fill_pv(pv1a, 3, 0);
  fill_pv(pv1b, 17, 0);
  fill_pv(pv2, 31, 1);
  // sa1: 55..62, sa2: 63..70
  const float* sa1c0b = IN(55); const float* sa1c0w = IN(56);
  const float* sa1n0b = IN(57); const float* sa1n0g = IN(58);
  const float* sa1c1b = IN(59); const float* sa1c1w = IN(60);
  const float* sa1n1b = IN(61); const float* sa1n1g = IN(62);
  const float* sa2c0b = IN(63); const float* sa2c0w = IN(64);
  const float* sa2n0b = IN(65); const float* sa2n0g = IN(66);
  const float* sa2c1b = IN(67); const float* sa2c1w = IN(68);
  const float* sa2n1b = IN(69); const float* sa2n1g = IN(70);

  // ---- workspace layout (floats) ----
  float* W = (float*)d_ws;
  size_t off = 0;
  auto alloc = [&](size_t n) { float* p = W + off; off += n; return p; };
  float* coords  = alloc((size_t)NB * 3 * 2048);
  float* featA   = alloc((size_t)NB * 32 * 2048);
  float* featB   = alloc((size_t)NB * 32 * 2048);
  float* featC   = alloc((size_t)NB * 32 * 1024);
  float* featD   = alloc((size_t)NB * 128 * 256);
  float* vmean   = alloc(16);
  float* vmaxb   = alloc(16);
  float* normb   = alloc((size_t)NB * 3 * 2048);
  float* vidxb   = alloc((size_t)NB * 2048);        // int
  const size_t GSZ = (size_t)NB * 35 * 1024 * 32;    // 4,587,520 floats each
  float* G0 = alloc(GSZ); float* G1 = alloc(GSZ); float* G2 = alloc(GSZ);
  float* cntb    = alloc((size_t)NB * 32768);
  float* svb     = alloc(128);
  float* sscl    = alloc(128);
  float* voxb    = alloc((size_t)NB * 32 * 2048);
  float* ptb     = alloc((size_t)NB * 32 * 2048);
  float* idxcb   = alloc((size_t)NB * 1024);         // int
  float* nidxb   = alloc((size_t)NB * 1024 * 32);    // int
  float* fmaxb   = alloc(512);
  float* cent1   = alloc((size_t)NB * 3 * 1024);
  float* cent2   = alloc((size_t)NB * 3 * 256);

  auto zero = [&](float* p, int n) { k_zero<<<dim3((n + 255) / 256), 256, 0, stream>>>(p, n); };

  // ---- coords = x.transpose(0,2,1) ----
  k_transpose<<<dim3((NB * 2048 + 255) / 256), 256, 0, stream>>>(X, coords, 2048, NB * 2048);

  auto pvconv = [&](const float* feat, int Cf, const float* crd, int N, int r,
                    const PVP& p, float* outfeat) {
    int V = r * r * r;
    k_vox_stats<<<NB, 256, 0, stream>>>(crd, N, vmean, vmaxb);
    k_vox_norm<<<dim3((NB * N + 255) / 256), 256, 0, stream>>>(crd, vmean, vmaxb, N, r,
                                                               normb, (int*)vidxb, NB * N);
    zero(G0, NB * Cf * V);
    zero(cntb, NB * V);
    k_scatter<<<dim3((NB * N + 255) / 256), 256, 0, stream>>>(feat, (int*)vidxb, Cf, N, V,
                                                              G0, cntb, NB * N);
    k_vox_div<<<dim3((NB * Cf * V + 255) / 256), 256, 0, stream>>>(G0, cntb, Cf, V, NB * Cf * V);
    if (Cf == 3)
      k_conv3d_t<3><<<dim3(V / 16, 2, NB), 32, 0, stream>>>(G0, p.c1w, p.c1b, G1, r);
    else
      k_conv3d_t<32><<<dim3(V / 16, 2, NB), 32, 0, stream>>>(G0, p.c1w, p.c1b, G1, r);
    k_gn_act<<<NB * 8, 256, 0, stream>>>(G1, p.n1g, p.n1b, 32, V, 1);
    k_conv3d_t<32><<<dim3(V / 16, 2, NB), 32, 0, stream>>>(G1, p.c2w, p.c2b, G2, r);
    k_gn_act<<<NB * 8, 256, 0, stream>>>(G2, p.n2g, p.n2b, 32, V, p.att ? 0 : 1);
    float* gcur = G2;
    if (p.att) {
      size_t BCV = (size_t)NB * 32 * V;
      float* Q = G0; float* K = G0 + BCV; float* Vv = G0 + 2 * BCV; float* HH = G0 + 3 * BCV;
      launch_gemm(32, 32, G2, p.aqw, p.aqb, nullptr, Q, V, stream);
      launch_gemm(32, 32, G2, p.akw, p.akb, nullptr, K, V, stream);
      launch_gemm(32, 32, G2, p.avw, p.avb, nullptr, Vv, V, stream);
      k_flash<<<dim3(V / 16, 1, NB), 32, 0, stream>>>(Q, K, Vv, HH, V);
      launch_gemm(32, 32, HH, p.aow, p.aob, G2, G1, V, stream);  // residual folded in
      k_gn_act<<<NB * 8, 256, 0, stream>>>(G1, p.ang, p.anb, 32, V, 1);
      gcur = G1;
    }
    k_se_mean<<<NB * 32, 256, 0, stream>>>(gcur, V, svb);
    k_se_mlp<<<NB, 32, 0, stream>>>(svb, p.se1, p.se2, sscl);
    k_se_scale<<<dim3((NB * 32 * V + 255) / 256), 256, 0, stream>>>(gcur, sscl, V, NB * 32 * V);
    k_devox<<<dim3((NB * N + 255) / 256), 256, 0, stream>>>(gcur, normb, N, r, 32, voxb, NB * N);
    launch_gemm(Cf, 32, feat, p.ptcw, p.ptcb, nullptr, ptb, N, stream);
    k_gn_act<<<NB * 8, 256, 0, stream>>>(ptb, p.ptng, p.ptnb, 32, N, 1);
    k_add<<<dim3((NB * 32 * N + 255) / 256), 256, 0, stream>>>(voxb, ptb, outfeat, NB * 32 * N);
  };

  auto sa = [&](const float* feat, const float* crd, int N, int m, float rad, int kN,
                const float* c0w, const float* c0b, const float* n0g, const float* n0b,
                const float* c1w, const float* c1b, const float* n1g, const float* n1b,
                int C2, float* outfeat, float* outcent) {
    k_fps<<<NB, 256, 0, stream>>>(crd, N, m, (int*)idxcb);
    k_centers<<<dim3((NB * m + 255) / 256), 256, 0, stream>>>(crd, (int*)idxcb, N, m,
                                                              outcent, NB * m);
    k_ballq<<<dim3((NB * m + 255) / 256), 256, 0, stream>>>(outcent, crd, N, m, rad, kN,
                                                            (int*)nidxb, NB * m);
    int S = m * kN;
    k_group<<<dim3((NB * S + 255) / 256), 256, 0, stream>>>(crd, feat, outcent, (int*)nidxb,
                                                            N, m, kN, G0, NB * S);
    launch_gemm(35, 32, G0, c0w, c0b, nullptr, G1, S, stream);
    k_gn_act<<<NB * 8, 256, 0, stream>>>(G1, n0g, n0b, 32, S, 1);
    launch_gemm(32, C2, G1, c1w, c1b, nullptr, G0, S, stream);
    k_gn_act<<<NB * 8, 256, 0, stream>>>(G0, n1g, n1b, C2, S, 1);
    k_maxlast<<<dim3((NB * C2 * m + 255) / 256), 256, 0, stream>>>(G0, kN, outfeat, NB * C2 * m);
  };

  // ---- pipeline ----
  pvconv(coords, 3, coords, 2048, 32, pv1a, featA);
  pvconv(featA, 32, coords, 2048, 32, pv1b, featB);
  sa(featB, coords, 2048, 1024, 0.1f, 32,
     sa1c0w, sa1c0b, sa1n0g, sa1n0b, sa1c1w, sa1c1b, sa1n1g, sa1n1b, 32, featC, cent1);
  pvconv(featC, 32, cent1, 1024, 16, pv2, featA);
  sa(featA, cent1, 1024, 256, 0.2f, 32,
     sa2c0w, sa2c0b, sa2n0g, sa2n0b, sa2c1w, sa2c1b, sa2n1g, sa2n1b, 128, featD, cent2);
  k_maxlast<<<dim3((NB * 128 + 255) / 256), 256, 0, stream>>>(featD, 256, fmaxb, NB * 128);
  k_head<<<dim3((NB * 256 + 255) / 256), 256, 0, stream>>>(fmaxb, head_w, head_b, (float*)d_out);
#undef IN
}
